// Attention_88673894793650
// MI455X (gfx1250) — matrix operand order repaired and checked by reading, not compiled, not run on hardware
//
#include <hip/hip_runtime.h>
#include <hip/hip_bf16.h>

typedef __attribute__((ext_vector_type(16))) _Float16 v16h;
typedef __attribute__((ext_vector_type(8)))  float    v8f;

#define DIM   768
#define HD    64
#define NH    12
#define LSEQ  1024
#define BATCH 4

union FragA {
    v16h     v;
    _Float16 f[16];
    unsigned u[8];
};

static __device__ inline v8f zero8() {
    v8f z = {0.f,0.f,0.f,0.f,0.f,0.f,0.f,0.f};
    return z;
}

static __device__ inline v8f wmma_f16(v16h a, v16h b, v8f c) {
    // 8 args: (neg_a, A, neg_b, B, c_mod, C, reuse_a, reuse_b)
    return __builtin_amdgcn_wmma_f32_16x16x32_f16(false, a, false, b, (short)0, c, false, false);
}

// ---------------------------------------------------------------------------
// GEMM: C[M,N] = A[M,K] @ W[N,K]^T  (K = N = DIM, M = 4096)
//  A_F16   : 0 = A is f32, 1 = A is f16
//  OUT_MODE: 0 = f16 row-major [M,DIM]
//            1 = f16 per-head transposed vT[b][h][d][key]
//            2 = f32 row-major [M,DIM] + bias
// block = 128 threads (4 waves); block tile 64(M) x 64(N); wave tile 16 x 64
// ---------------------------------------------------------------------------
template<int A_F16, int OUT_MODE>
__global__ __launch_bounds__(128) void gemm_wmma_k(
    const void* __restrict__ Ap, const float* __restrict__ W,
    const float* __restrict__ bias, void* __restrict__ Out)
{
    const int lane  = threadIdx.x & 31;
    const int wave  = threadIdx.x >> 5;
    const int row16 = lane & 15;
    const int halfs = lane >> 4;
    const int m0 = blockIdx.x * 64 + wave * 16;
    const int n0 = blockIdx.y * 64;
    const int m  = m0 + row16;

    const float*    Af = (const float*)Ap;
    const _Float16* Ah = (const _Float16*)Ap;

    v8f acc[4];
    #pragma unroll
    for (int t = 0; t < 4; ++t) acc[t] = zero8();

    for (int k0 = 0; k0 < DIM; k0 += 32) {
        // A fragment: element e -> K = e + 16*halfs (contiguous k); element pairs
        // (2g,2g+1) are contiguous K values => 32-bit loads (f16) / float2 loads (f32).
        FragA a;
        #pragma unroll
        for (int g = 0; g < 8; ++g) {
            int kk = k0 + 2*g + 16*halfs;
            if (A_F16) {
                a.u[g] = *(const unsigned*)(Ah + (size_t)m*DIM + kk);
            } else {
                float2 f2 = *(const float2*)(Af + (size_t)m*DIM + kk);
                a.f[2*g]   = (_Float16)f2.x;
                a.f[2*g+1] = (_Float16)f2.y;
            }
        }
        #pragma unroll
        for (int t = 0; t < 4; ++t) {
            // B[k][n] = W[n][k]: element e -> K = e + 16*halfs (contiguous k)
            int n = n0 + t*16 + row16;
            const float4* wp = (const float4*)(W + (size_t)n*DIM + k0 + 16*halfs);
            FragA bf;
            #pragma unroll
            for (int q4 = 0; q4 < 4; ++q4) {
                float4 f = wp[q4];
                bf.f[4*q4+0] = (_Float16)f.x;
                bf.f[4*q4+1] = (_Float16)f.y;
                bf.f[4*q4+2] = (_Float16)f.z;
                bf.f[4*q4+3] = (_Float16)f.w;
            }
            acc[t] = wmma_f16(a.v, bf.v, acc[t]);
        }
    }

    #pragma unroll
    for (int t = 0; t < 4; ++t) {
        #pragma unroll
        for (int r = 0; r < 8; ++r) {
            int mm = m0 + r + 8*halfs;        // C layout: row = r + 8*(lane/16)
            int nn = n0 + t*16 + row16;       //           col = lane%16
            float val = acc[t][r];
            if (OUT_MODE == 2) {
                ((float*)Out)[(size_t)mm*DIM + nn] = val + bias[nn];
            } else if (OUT_MODE == 0) {
                ((_Float16*)Out)[(size_t)mm*DIM + nn] = (_Float16)val;
            } else {
                int bb = mm >> 10, key = mm & 1023;
                int hh = nn >> 6,  d   = nn & 63;
                ((_Float16*)Out)[((size_t)((bb*NH + hh)*HD + d) << 10) + key] = (_Float16)val;
            }
        }
    }
}

// ---------------------------------------------------------------------------
// Attention: one block = 64 threads (2 waves); wave owns 16 q rows of (b,h).
// Full 16x1024 score strip in LDS (pad 1028 -> conflict-free fragment reads).
// ---------------------------------------------------------------------------
__global__ __launch_bounds__(64) void attn_k(
    const _Float16* __restrict__ qf, const _Float16* __restrict__ kf,
    const _Float16* __restrict__ vT, const float* __restrict__ relk,
    const float* __restrict__ relv, float* __restrict__ attn_out,
    _Float16* __restrict__ wbuf)
{
    __shared__ float S[2][16][1028];
    __shared__ float qrelS[2][16][8];
    __shared__ float bsum[2][16][8];
    __shared__ float rowinv[2][16];

    const int lane  = threadIdx.x & 31;
    const int w     = threadIdx.x >> 5;
    const int row16 = lane & 15;
    const int halfs = lane >> 4;

    const int blk = blockIdx.x;                 // B*NH*32 blocks
    const int qb  = blk & 31;
    const int h   = (blk >> 5) % NH;
    const int b   = blk / (32 * NH);
    const int qbase = qb*32 + w*16;

    const size_t bh = (size_t)(b*NH + h);
    const _Float16* qrow = qf + ((size_t)b*LSEQ)*DIM + h*HD;   // + q*DIM + d
    const _Float16* krow = kf + ((size_t)b*LSEQ)*DIM + h*HD;   // + key*DIM + d
    const _Float16* vrow = vT + (bh*HD << 10);                 // + d*1024 + key

    // --- qrel[m][j] = q[m] . rel_k_table[j]   (112 dot products / wave)
    for (int t = lane; t < 112; t += 32) {
        int mm = t / 7, j = t % 7;
        const _Float16* qp = qrow + (size_t)(qbase + mm)*DIM;
        const float*    rp = relk + j*HD;
        float s = 0.f;
        for (int d = 0; d < HD; ++d) s += (float)qp[d] * rp[d];
        qrelS[w][mm][j] = s;
    }

    // --- Q A-fragments (two K-slices of hd=64), reused for all 64 key tiles
    FragA aQ[2];
    #pragma unroll
    for (int sl = 0; sl < 2; ++sl)
        #pragma unroll
        for (int g = 0; g < 8; ++g) {
            int d = sl*32 + 2*g + 16*halfs;
            aQ[sl].u[g] = *(const unsigned*)(qrow + (size_t)(qbase + row16)*DIM + d);
        }

    // --- S = (Q K^T + rel) * sqrt(hd)   -> LDS
    for (int kt = 0; kt < 64; ++kt) {
        v8f acc = zero8();
        const int key = kt*16 + row16;          // B col index = lane%16
        #pragma unroll
        for (int sl = 0; sl < 2; ++sl) {
            v16h bk = *(const v16h*)(krow + (size_t)key*DIM + sl*32 + 16*halfs);
            acc = wmma_f16(aQ[sl].v, bk, acc);
        }
        #pragma unroll
        for (int r = 0; r < 8; ++r) {
            int mm   = r + 8*halfs;
            int qg   = qbase + mm;
            int kcol = kt*16 + row16;
            int dv   = kcol - qg;
            int j    = min(max(dv, -3), 3) + 3;
            S[w][mm][kcol] = (acc[r] + qrelS[w][mm][j]) * 8.0f;  // /scale = *sqrt(64)
        }
    }

    // --- per-row softmax + bucket sums + single global write of attn
    for (int mm = 0; mm < 16; ++mm) {
        const int qg = qbase + mm;
        float mx = -3.4e38f;
        for (int k = lane; k < 1024; k += 32) mx = fmaxf(mx, S[w][mm][k]);
        #pragma unroll
        for (int off = 16; off > 0; off >>= 1) mx = fmaxf(mx, __shfl_xor(mx, off, 32));

        float sum = 0.f, blo = 0.f, bhi = 0.f;
        for (int k = lane; k < 1024; k += 32) {
            float e = __expf(S[w][mm][k] - mx);
            S[w][mm][k] = e;
            sum += e;
            if (k <= qg - 3) blo += e;
            if (k >= qg + 3) bhi += e;
        }
        #pragma unroll
        for (int off = 16; off > 0; off >>= 1) {
            sum += __shfl_xor(sum, off, 32);
            blo += __shfl_xor(blo, off, 32);
            bhi += __shfl_xor(bhi, off, 32);
        }
        float inv = 1.0f / sum;
        if (lane == 0) {
            rowinv[w][mm]  = inv;
            bsum[w][mm][0] = blo;
            bsum[w][mm][6] = bhi;
        } else if (lane <= 5) {                 // j = 1..5 <-> k-qg = -2..2
            int k = qg + lane - 3;
            bsum[w][mm][lane] = (k >= 0 && k < 1024) ? S[w][mm][k] : 0.f;
        }
        // normalize in LDS and write attn (float4 per lane)
        float* arow = attn_out + ((bh << 10) + (size_t)qg) * 1024;
        for (int k0 = 0; k0 < 1024; k0 += 128) {
            int k = k0 + lane*4;
            float4 p;
            p.x = S[w][mm][k]   * inv;
            p.y = S[w][mm][k+1] * inv;
            p.z = S[w][mm][k+2] * inv;
            p.w = S[w][mm][k+3] * inv;
            S[w][mm][k] = p.x; S[w][mm][k+1] = p.y;
            S[w][mm][k+2] = p.z; S[w][mm][k+3] = p.w;
            *(float4*)(arow + k) = p;
        }
    }

    // --- w1 = P @ V   (A from LDS, B contiguous from vT)
    v8f accw[4];
    #pragma unroll
    for (int t = 0; t < 4; ++t) accw[t] = zero8();

    for (int step = 0; step < 32; ++step) {
        FragA aP;
        #pragma unroll
        for (int g = 0; g < 8; ++g)
            #pragma unroll
            for (int r = 0; r < 2; ++r) {
                int kk = step*32 + 2*g + 16*halfs + r;
                aP.f[2*g + r] = (_Float16)S[w][row16][kk];
            }
        #pragma unroll
        for (int t = 0; t < 4; ++t) {
            int d = t*16 + row16;
            v16h bv = *(const v16h*)(vrow + (size_t)d*1024 + step*32 + 16*halfs);
            accw[t] = wmma_f16(aP.v, bv, accw[t]);
        }
    }

    // --- w = w1 + w2, stored f16 for the final projection
    #pragma unroll
    for (int t = 0; t < 4; ++t)
        #pragma unroll
        for (int r = 0; r < 8; ++r) {
            int mm = r + 8*halfs;
            int qg = qbase + mm;
            int d  = t*16 + row16;
            float w2 = 0.f;
            #pragma unroll
            for (int j = 0; j < 7; ++j) w2 += bsum[w][mm][j] * relv[j*HD + d];
            float val = accw[t][r] + w2 * rowinv[w][mm];
            wbuf[((size_t)(b*LSEQ + qg))*DIM + h*HD + d] = (_Float16)val;
        }
}

// ---------------------------------------------------------------------------
extern "C" void kernel_launch(void* const* d_in, const int* in_sizes, int n_in,
                              void* d_out, int out_size, void* d_ws, size_t ws_size,
                              hipStream_t stream) {
    (void)in_sizes; (void)n_in; (void)out_size; (void)ws_size;

    const float* query = (const float*)d_in[0];
    const float* key   = (const float*)d_in[1];
    const float* value = (const float*)d_in[2];
    const float* Wq    = (const float*)d_in[3];
    const float* Wk    = (const float*)d_in[4];
    const float* Wv    = (const float*)d_in[5];
    const float* Wproj = (const float*)d_in[6];
    const float* bproj = (const float*)d_in[7];
    const float* relk  = (const float*)d_in[8];
    const float* relv  = (const float*)d_in[9];

    const size_t NELEM = (size_t)BATCH * LSEQ * DIM;   // 3,145,728
    _Float16* qf = (_Float16*)d_ws;
    _Float16* kf = qf + NELEM;
    _Float16* vT = kf + NELEM;
    _Float16* wb = vT + NELEM;

    float* x_out    = (float*)d_out;
    float* attn_out = x_out + NELEM;

    dim3 gg(BATCH * LSEQ / 64, DIM / 64);   // 64 x 12
    gemm_wmma_k<0,0><<<gg, 128, 0, stream>>>(query, Wq, nullptr, qf);
    gemm_wmma_k<0,0><<<gg, 128, 0, stream>>>(key,   Wk, nullptr, kf);
    gemm_wmma_k<0,1><<<gg, 128, 0, stream>>>(value, Wv, nullptr, vT);

    attn_k<<<BATCH * NH * 32, 64, 0, stream>>>(qf, kf, vT, relk, relv, attn_out, wb);

    gemm_wmma_k<1,2><<<gg, 128, 0, stream>>>(wb, Wproj, bproj, x_out);
}
